// MemoryAugmentedNetwork_14955076125123
// MI455X (gfx1250) — compile-verified, matrix operand order checked by reading, not run on hardware
//
#include <hip/hip_runtime.h>
#include <hip/hip_bf16.h>
#include <stdint.h>

// ---------------------------------------------------------------------------
// Problem dims (fixed by the reference)
// ---------------------------------------------------------------------------
#define B_   4
#define S_   256
#define IN_  256
#define H_   512
#define D_   128
#define M_   100000
#define OUT_ 256
#define BS_  1024          // B*S
#define G4H_ 2048          // 4*H
#define KCH_ 128           // keys per sims chunk
#define NCH_ ((M_ + KCH_ - 1) / KCH_)

typedef __attribute__((ext_vector_type(16))) __bf16 v16bf;
typedef __attribute__((ext_vector_type(8)))  __bf16 v8bf;
typedef __attribute__((ext_vector_type(8)))  float  v8f;
typedef __attribute__((ext_vector_type(4)))  unsigned int u32x4;
typedef __attribute__((ext_vector_type(8)))  unsigned int u32x8;

// ---------------------------------------------------------------------------
// WMMA helpers (CDNA5 wave32, v_wmma_f32_16x16x32_bf16)
// ---------------------------------------------------------------------------
__device__ __forceinline__ v16bf frag_from_f32(const float* __restrict__ p, int kb) {
  float4 a0 = *(const float4*)(p + kb);
  float4 a1 = *(const float4*)(p + kb + 4);
  float4 a2 = *(const float4*)(p + kb + 16);
  float4 a3 = *(const float4*)(p + kb + 20);
  v16bf v;
  v[0]=(__bf16)a0.x; v[1]=(__bf16)a0.y; v[2]=(__bf16)a0.z; v[3]=(__bf16)a0.w;
  v[4]=(__bf16)a1.x; v[5]=(__bf16)a1.y; v[6]=(__bf16)a1.z; v[7]=(__bf16)a1.w;
  v[8]=(__bf16)a2.x; v[9]=(__bf16)a2.y; v[10]=(__bf16)a2.z; v[11]=(__bf16)a2.w;
  v[12]=(__bf16)a3.x; v[13]=(__bf16)a3.y; v[14]=(__bf16)a3.z; v[15]=(__bf16)a3.w;
  return v;
}

__device__ __forceinline__ v16bf frag_from_bf16(const __bf16* p, int kb) {
  v8bf lo = *(const v8bf*)(p + kb);
  v8bf hi = *(const v8bf*)(p + kb + 16);
  return __builtin_shufflevector(lo, hi, 0,1,2,3,4,5,6,7,8,9,10,11,12,13,14,15);
}

__device__ __forceinline__ v8f wmma_bf16(v16bf a, v16bf b, v8f c) {
  return __builtin_amdgcn_wmma_f32_16x16x32_bf16(false, a, false, b, (short)0, c,
                                                 false, false);
}

// ---------------------------------------------------------------------------
// TDM: issue tensor_load_to_lds of a 2-D bf16 tile (tile_dim1 rows x 128 elems)
// D# per CDNA5 ISA.  Groups 2/3 = NULL (2-D tensor).
// ---------------------------------------------------------------------------
__device__ __forceinline__ void tdm_load_keys(const __bf16* gsrc, uint32_t lds_off,
                                              uint32_t rows_remaining) {
  uint64_t ga = (uint64_t)(uintptr_t)gsrc;
  u32x4 g0;
  g0[0] = 1u;                                              // count=1, user mode
  g0[1] = __builtin_amdgcn_readfirstlane((int)lds_off);    // lds_addr (bytes)
  g0[2] = (uint32_t)ga;                                    // global_addr[31:0]
  g0[3] = (uint32_t)((ga >> 32) & 0x1FFFFFFu) | (2u << 30);// addr[56:32] | type=2
  u32x8 g1;
  g1[0] = 0x00010000u;                                     // data_size=1 (2 bytes)
  g1[1] = (uint32_t)D_ << 16;                              // tensor_dim0 lo16 = 128
  g1[2] = (rows_remaining & 0xffffu) << 16;                // tdim0 hi=0 | tdim1 lo16
  g1[3] = ((rows_remaining >> 16) & 0xffffu)               // tdim1 hi16
          | ((uint32_t)D_ << 16);                          // tile_dim0 = 128
  g1[4] = (uint32_t)KCH_;                                  // tile_dim1=128, tile_dim2=0
  g1[5] = (uint32_t)D_;                                    // tensor_dim0_stride = 128
  g1[6] = 0u;
  g1[7] = 0u;
  asm volatile("tensor_load_to_lds %0, %1" :: "s"(g0), "s"(g1) : "memory");
}

__device__ __forceinline__ void wait_tensorcnt_0() {
#if __has_builtin(__builtin_amdgcn_s_wait_tensorcnt)
  __builtin_amdgcn_s_wait_tensorcnt((short)0);
#else
  asm volatile("s_wait_tensorcnt 0" ::: "memory");
#endif
}
__device__ __forceinline__ void wait_tensorcnt_1() {
#if __has_builtin(__builtin_amdgcn_s_wait_tensorcnt)
  __builtin_amdgcn_s_wait_tensorcnt((short)1);
#else
  asm volatile("s_wait_tensorcnt 1" ::: "memory");
#endif
}

// ---------------------------------------------------------------------------
// Generic GEMM: C[M x N] = A[M x K] (fp32, row-major) * W[N x K]^T + bias1+bias2
// block = 256 threads (8 waves); each wave computes one 16x16 tile.
// grid = ((N+127)/128, M/16).  K is compile-time -> fully unrolled WMMA stream.
// ---------------------------------------------------------------------------
template <int K>
__global__ void k_gemm_bf16(const float* __restrict__ A, const float* __restrict__ W,
                            const float* __restrict__ bias1, const float* __restrict__ bias2,
                            float* __restrict__ C, int N) {
  int lane = threadIdx.x & 31;
  int wave = threadIdx.x >> 5;
  int ntile = blockIdx.x * 8 + wave;
  if (ntile * 16 >= N) return;
  int mtile = blockIdx.y;

  const float* arow = A + (size_t)(mtile * 16 + (lane & 15)) * K;
  const float* brow = W + (size_t)(ntile * 16 + (lane & 15)) * K;
  int koff = (lane & 16) ? 8 : 0;

  v8f acc = {0.f,0.f,0.f,0.f,0.f,0.f,0.f,0.f};
  #pragma unroll
  for (int k0 = 0; k0 < K; k0 += 32) {
    v16bf af = frag_from_f32(arow, k0 + koff);
    v16bf bf = frag_from_f32(brow, k0 + koff);
    acc = wmma_bf16(af, bf, acc);
  }

  int col = ntile * 16 + (lane & 15);
  float badd = 0.f;
  if (bias1) badd += bias1[col];
  if (bias2) badd += bias2[col];
  int rbase = mtile * 16 + ((lane & 16) ? 8 : 0);
  #pragma unroll
  for (int j = 0; j < 8; ++j)
    C[(size_t)(rbase + j) * N + col] = acc[j] + badd;
}

// ---------------------------------------------------------------------------
// LSTM recurrence: one block per batch element, 512 threads; h in LDS, c in reg.
// ---------------------------------------------------------------------------
__device__ __forceinline__ float sigmoidf(float x) { return 1.f / (1.f + __expf(-x)); }

__global__ void k_lstm(const float* __restrict__ xg, const float* __restrict__ W_hh,
                       float* __restrict__ hs) {
  __shared__ float h_s[H_];
  int t = threadIdx.x;
  int b = blockIdx.x;
  h_s[t] = 0.f;
  float c = 0.f;
  __syncthreads();

  const float* Wi = W_hh + (size_t)t * H_;
  const float* Wf = W_hh + (size_t)(t + H_) * H_;
  const float* Wg = W_hh + (size_t)(t + 2 * H_) * H_;
  const float* Wo = W_hh + (size_t)(t + 3 * H_) * H_;

  for (int s = 0; s < S_; ++s) {
    const float* xr = xg + ((size_t)b * S_ + s) * G4H_;
    float gi = xr[t], gf = xr[t + H_], gg = xr[t + 2 * H_], go = xr[t + 3 * H_];
    for (int k = 0; k < H_; k += 4) {
      float4 hv = *(const float4*)(h_s + k);
      float4 w0 = *(const float4*)(Wi + k);
      float4 w1 = *(const float4*)(Wf + k);
      float4 w2 = *(const float4*)(Wg + k);
      float4 w3 = *(const float4*)(Wo + k);
      gi += hv.x*w0.x + hv.y*w0.y + hv.z*w0.z + hv.w*w0.w;
      gf += hv.x*w1.x + hv.y*w1.y + hv.z*w1.z + hv.w*w1.w;
      gg += hv.x*w2.x + hv.y*w2.y + hv.z*w2.z + hv.w*w2.w;
      go += hv.x*w3.x + hv.y*w3.y + hv.z*w3.z + hv.w*w3.w;
    }
    c = sigmoidf(gf) * c + sigmoidf(gi) * tanhf(gg);
    float h = sigmoidf(go) * tanhf(c);
    __syncthreads();
    h_s[t] = h;
    hs[((size_t)b * S_ + s) * H_ + t] = h;
    __syncthreads();
  }
}

// ---------------------------------------------------------------------------
// Row L2-normalize fp32 -> bf16.  One wave per row (C = 128).
// ---------------------------------------------------------------------------
__global__ void k_norm_rows(const float* __restrict__ in, __bf16* __restrict__ out, int R) {
  int lane = threadIdx.x & 31, wave = threadIdx.x >> 5;
  int row = blockIdx.x * (blockDim.x >> 5) + wave;
  if (row >= R) return;
  const float* p = in + (size_t)row * D_;
  float v0 = p[lane], v1 = p[lane + 32], v2 = p[lane + 64], v3 = p[lane + 96];
  float ss = v0*v0 + v1*v1 + v2*v2 + v3*v3;
  #pragma unroll
  for (int o = 16; o; o >>= 1) ss += __shfl_xor(ss, o, 32);
  float inv = 1.0f / fmaxf(sqrtf(ss), 1e-12f);
  __bf16* q = out + (size_t)row * D_;
  q[lane]      = (__bf16)(v0 * inv);
  q[lane + 32] = (__bf16)(v1 * inv);
  q[lane + 64] = (__bf16)(v2 * inv);
  q[lane + 96] = (__bf16)(v3 * inv);
}

// ---------------------------------------------------------------------------
// Branchless top-3 insert (v_cmp + v_cndmask, no exec-mask churn).
// Invariant t0 >= t1 >= t2.
// ---------------------------------------------------------------------------
__device__ __forceinline__ void top3_ins(float v, int ii,
                                         float& t0, float& t1, float& t2,
                                         int& i0, int& i1, int& i2) {
  bool g0 = v > t0, g1 = v > t1, g2 = v > t2;
  t2 = g1 ? t1 : (g2 ? v : t2);  i2 = g1 ? i1 : (g2 ? ii : i2);
  t1 = g0 ? t0 : (g1 ? v : t1);  i1 = g0 ? i0 : (g1 ? ii : i1);
  t0 = g0 ? v : t0;              i0 = g0 ? ii : i0;
}

// ---------------------------------------------------------------------------
// Fused cosine sims + top-3.  Block = 256 threads (8 waves) owns 16 queries.
// TDM double-buffers 128-key x 128-elem bf16 tiles into LDS; 8 waves each
// WMMA a 16x16 sims tile into LDS; 256 threads scan (16 queries x 16 groups
// of 8 keys) with branchless top-3; hierarchical merge at the end.
// Query tile staged via async global->LDS load.
// ---------------------------------------------------------------------------
__global__ void k_sims_topk(const __bf16* __restrict__ qn, const __bf16* __restrict__ kn,
                            int* __restrict__ topi) {
  __shared__ __bf16 kbuf[2 * KCH_ * D_];     // 64 KB double-buffered key tiles
  __shared__ __bf16 qtile[16 * D_];          // 4 KB
  __shared__ float  tile[16][132];           // 16 queries x 128 keys (+pad)
  __shared__ float  pv[16][16][3];
  __shared__ int    pi[16][16][3];

  int tid = threadIdx.x, lane = tid & 31, wave = tid >> 5;
  int qbase = blockIdx.x * 16;

  // --- TDM: kick off key chunk 0 while we stage queries -------------------
  if (wave == 0)
    tdm_load_keys(kn, (uint32_t)(uintptr_t)kbuf, (uint32_t)M_);

  // --- async-stage the 16x128 bf16 query tile into LDS (16 B per thread) ---
  {
    uint64_t g = (uint64_t)(uintptr_t)(qn + (size_t)qbase * D_ + tid * 8);
    uint32_t l = (uint32_t)(uintptr_t)(qtile + tid * 8);
    asm volatile("global_load_async_to_lds_b128 %0, %1, off" :: "v"(l), "v"(g) : "memory");
    asm volatile("s_wait_asynccnt 0" ::: "memory");
  }
  __syncthreads();

  int koff = (lane & 16) ? 8 : 0;
  v16bf afrag[4];
  {
    const __bf16* ap = qtile + (size_t)(lane & 15) * D_;
    #pragma unroll
    for (int ks = 0; ks < 4; ++ks) afrag[ks] = frag_from_bf16(ap, ks * 32 + koff);
  }

  int qloc = tid & 15, grp = tid >> 4;       // 16 groups x 8 keys each
  float t0 = -1e30f, t1 = -1e30f, t2 = -1e30f;
  int   i0 = 0, i1 = 0, i2 = 0;

  for (int c = 0; c < NCH_; ++c) {
    int base = c * KCH_;
    if (wave == 0) {
      if (c + 1 < NCH_) {       // prefetch next chunk into other buffer
        tdm_load_keys(kn + (size_t)(base + KCH_) * D_,
                      (uint32_t)(uintptr_t)(kbuf + ((c + 1) & 1) * (KCH_ * D_)),
                      (uint32_t)(M_ - (base + KCH_)));
        wait_tensorcnt_1();     // chunk c complete (in-order)
      } else {
        wait_tensorcnt_0();
      }
    }
    __syncthreads();            // chunk c visible to all waves

    // -- compute 16x16 sims tile per wave: load all B-frags, then WMMA ----
    const __bf16* bp = kbuf + (size_t)(c & 1) * (KCH_ * D_)
                            + (size_t)(wave * 16 + (lane & 15)) * D_;
    v16bf bfr[4];
    #pragma unroll
    for (int ks = 0; ks < 4; ++ks) bfr[ks] = frag_from_bf16(bp, ks * 32 + koff);
    v8f acc = {0.f,0.f,0.f,0.f,0.f,0.f,0.f,0.f};
    #pragma unroll
    for (int ks = 0; ks < 4; ++ks) acc = wmma_bf16(afrag[ks], bfr[ks], acc);

    int col = wave * 16 + (lane & 15);
    int rb  = (lane & 16) ? 8 : 0;
    #pragma unroll
    for (int j = 0; j < 8; ++j) tile[rb + j][col] = acc[j];
    __syncthreads();            // sims tile ready

    // -- branchless top-3 scan: 8 keys per thread as two float4 loads -----
    {
      const float* trow = &tile[qloc][grp * 8];
      float4 s0 = *(const float4*)(trow);
      float4 s1 = *(const float4*)(trow + 4);
      int kk = base + grp * 8;
      float vv[8] = {s0.x, s0.y, s0.z, s0.w, s1.x, s1.y, s1.z, s1.w};
      #pragma unroll
      for (int i = 0; i < 8; ++i) {
        float v = (kk + i < M_) ? vv[i] : -1e30f;
        top3_ins(v, kk + i, t0, t1, t2, i0, i1, i2);
      }
    }
    __syncthreads();            // tile consumed
  }

  pv[qloc][grp][0] = t0; pv[qloc][grp][1] = t1; pv[qloc][grp][2] = t2;
  pi[qloc][grp][0] = i0; pi[qloc][grp][1] = i1; pi[qloc][grp][2] = i2;
  __syncthreads();

  if (tid < 16) {
    float m0 = -1e30f, m1 = -1e30f, m2 = -1e30f; int j0 = 0, j1 = 0, j2 = 0;
    for (int g = 0; g < 16; ++g)
      #pragma unroll
      for (int k = 0; k < 3; ++k)
        top3_ins(pv[tid][g][k], pi[tid][g][k], m0, m1, m2, j0, j1, j2);
    int* o = topi + (size_t)(qbase + tid) * 3;
    o[0] = j0; o[1] = j1; o[2] = j2;
  }
}

// ---------------------------------------------------------------------------
// Gather retrieved values, attention softmax over K=3, weighted sum.
// One wave per token; prefetch the 3 gathered rows (global_prefetch_b8).
// ---------------------------------------------------------------------------
__global__ void k_attend(const float* __restrict__ vals, const int* __restrict__ topi,
                         const float* __restrict__ Wa, const float* __restrict__ ba,
                         float* __restrict__ memv) {
  int lane = threadIdx.x & 31, wave = threadIdx.x >> 5;
  int tok = blockIdx.x * (blockDim.x >> 5) + wave;
  if (tok >= BS_) return;

  int idx[3];
  #pragma unroll
  for (int k = 0; k < 3; ++k) {
    idx[k] = topi[tok * 3 + k];
    __builtin_prefetch((const void*)(vals + (size_t)idx[k] * D_), 0, 3);
  }

  float v[3][4];
  float logit[3];
  #pragma unroll
  for (int k = 0; k < 3; ++k) {
    const float* vp = vals + (size_t)idx[k] * D_;
    float p = 0.f;
    #pragma unroll
    for (int c = 0; c < 4; ++c) {
      float x = vp[lane + 32 * c];
      v[k][c] = x;
      p += x * Wa[lane + 32 * c];
    }
    #pragma unroll
    for (int o = 16; o; o >>= 1) p += __shfl_xor(p, o, 32);
    logit[k] = p + ba[0];
  }
  float mx = fmaxf(logit[0], fmaxf(logit[1], logit[2]));
  float e0 = __expf(logit[0] - mx), e1 = __expf(logit[1] - mx), e2 = __expf(logit[2] - mx);
  float inv = 1.f / (e0 + e1 + e2);
  #pragma unroll
  for (int c = 0; c < 4; ++c)
    memv[(size_t)tok * D_ + lane + 32 * c] = (e0 * v[0][c] + e1 * v[1][c] + e2 * v[2][c]) * inv;
}

// ---------------------------------------------------------------------------
// Concat u = [hs (BSxH), memc (BSxD)]  -> (BS x (H+D))
// ---------------------------------------------------------------------------
__global__ void k_concat(const float* __restrict__ hs, const float* __restrict__ memc,
                         float* __restrict__ u) {
  int i = blockIdx.x * blockDim.x + threadIdx.x;
  if (i >= BS_ * (H_ + D_)) return;
  int r = i / (H_ + D_), c = i - r * (H_ + D_);
  u[i] = (c < H_) ? hs[(size_t)r * H_ + c] : memc[(size_t)r * D_ + (c - H_)];
}

// ---------------------------------------------------------------------------
// Launch
// ---------------------------------------------------------------------------
extern "C" void kernel_launch(void* const* d_in, const int* in_sizes, int n_in,
                              void* d_out, int out_size, void* d_ws, size_t ws_size,
                              hipStream_t stream) {
  const float* x     = (const float*)d_in[0];
  const float* W_ih  = (const float*)d_in[1];
  const float* W_hh  = (const float*)d_in[2];
  const float* b_ih  = (const float*)d_in[3];
  const float* b_hh  = (const float*)d_in[4];
  const float* Wq    = (const float*)d_in[5];
  const float* bq    = (const float*)d_in[6];
  const float* Wa    = (const float*)d_in[7];
  const float* ba    = (const float*)d_in[8];
  const float* Wc    = (const float*)d_in[9];
  const float* bc    = (const float*)d_in[10];
  const float* Wo    = (const float*)d_in[11];
  const float* bo    = (const float*)d_in[12];
  const float* mkeys = (const float*)d_in[13];
  const float* mvals = (const float*)d_in[14];
  float* out = (float*)d_out;

  char* ws = (char*)d_ws;
  size_t off = 0;
  auto alloc = [&](size_t bytes) { char* p = ws + off; off += (bytes + 255) & ~(size_t)255; return p; };
  float*  xg     = (float*) alloc((size_t)BS_ * G4H_ * 4);   // 8 MB
  float*  hs     = (float*) alloc((size_t)BS_ * H_ * 4);     // 2 MB
  float*  q_raw  = (float*) alloc((size_t)BS_ * D_ * 4);
  __bf16* q_bf   = (__bf16*)alloc((size_t)BS_ * D_ * 2);
  __bf16* kn_bf  = (__bf16*)alloc((size_t)M_ * D_ * 2);      // 25.6 MB
  int*    topi   = (int*)   alloc((size_t)BS_ * 3 * 4);
  float*  memv   = (float*) alloc((size_t)BS_ * D_ * 4);
  float*  memc   = (float*) alloc((size_t)BS_ * D_ * 4);
  float*  u      = (float*) alloc((size_t)BS_ * (H_ + D_) * 4);
  (void)ws_size; (void)in_sizes; (void)n_in; (void)out_size;

  // 1) xg = x @ W_ih^T + (b_ih + b_hh)        (1024 x 2048, K=256)
  k_gemm_bf16<IN_><<<dim3(G4H_ / 128, BS_ / 16), 256, 0, stream>>>(x, W_ih, b_ih, b_hh, xg, G4H_);
  // 2) LSTM scan -> hs (1024 x 512)
  k_lstm<<<B_, H_, 0, stream>>>(xg, W_hh, hs);
  // 3) q_raw = hs @ Wq^T + bq                 (1024 x 128, K=512)
  k_gemm_bf16<H_><<<dim3(1, BS_ / 16), 256, 0, stream>>>(hs, Wq, bq, nullptr, q_raw, D_);
  // 4) normalize q and keys -> bf16
  k_norm_rows<<<BS_ / 8, 256, 0, stream>>>(q_raw, q_bf, BS_);
  k_norm_rows<<<M_ / 8, 256, 0, stream>>>(mkeys, kn_bf, M_);
  // 5) fused cosine sims + top-3 (TDM-streamed keys)
  k_sims_topk<<<BS_ / 16, 256, 0, stream>>>(q_bf, kn_bf, topi);
  // 6) gather + attention softmax + weighted sum -> memv
  k_attend<<<BS_ / 8, 256, 0, stream>>>(mvals, topi, Wa, ba, memv);
  // 7) memc = memv @ Wc^T + bc                (1024 x 128, K=128)
  k_gemm_bf16<D_><<<dim3(1, BS_ / 16), 256, 0, stream>>>(memv, Wc, bc, nullptr, memc, D_);
  // 8) u = concat(hs, memc)                   (1024 x 640)
  k_concat<<<(BS_ * (H_ + D_) + 255) / 256, 256, 0, stream>>>(hs, memc, u);
  // 9) out = u @ Wo^T + bo                    (1024 x 256, K=640)
  k_gemm_bf16<H_ + D_><<<dim3(OUT_ / 128, BS_ / 16), 256, 0, stream>>>(u, Wo, bo, nullptr, out, OUT_);
}
